// STGCN_28913719836924
// MI455X (gfx1250) — matrix operand order repaired and checked by reading, not compiled, not run on hardware
//
#include <hip/hip_runtime.h>

typedef __attribute__((ext_vector_type(16))) _Float16 v16h;
typedef __attribute__((ext_vector_type(8)))  float    v8f;

#define C_DIM 64

// ---------------- degree / dinv ----------------
__global__ void stgcn_init_deg(float* deg, int N) {
    int i = blockIdx.x * blockDim.x + threadIdx.x;
    if (i < N) deg[i] = 1.0f;
}

__global__ void stgcn_count_deg(const int* __restrict__ dst, float* deg, int E) {
    int e = blockIdx.x * blockDim.x + threadIdx.x;
    if (e < E) atomicAdd(&deg[dst[e]], 1.0f);
}

__global__ void stgcn_dinv(float* deg_dinv, int N) {
    int i = blockIdx.x * blockDim.x + threadIdx.x;
    if (i < N) deg_dinv[i] = rsqrtf(deg_dinv[i]);
}

// ---------------- GEMM: Y(N x 64) = X(N x 64) @ W(64 x 64) via WMMA ----------------
// Block = 256 threads = 8 waves; wave w -> m-subtile w>>2 (2 x 16 rows),
// n-subtile w&3 (4 x 16 cols); block covers 32 rows x 64 cols.
// W is repacked in LDS as k-pair dwords, n-major: sWp[n*32 + k/2] = {W[k][n], W[k+1][n]}
// so each B fragment is 2x ds_load_b128. A fragment is 4x global_load_b128 + cvt.
__global__ __launch_bounds__(256) void stgcn_gemm64_wmma(const float* __restrict__ X,
                                                         const float* __restrict__ W,
                                                         float* __restrict__ Y, int N) {
    __shared__ unsigned int sWp[C_DIM * 32];   // 8 KB: 64 cols x 32 k-pair dwords
    int t = threadIdx.x;
#pragma unroll
    for (int j = 0; j < 8; ++j) {
        int i = t * 8 + j;                 // word index 0..2047
        int n = i >> 5;                    // column of W
        int w = i & 31;                    // k-pair index
        union { unsigned u; _Float16 h[2]; } pk;
        pk.h[0] = (_Float16)W[(2 * w) * C_DIM + n];
        pk.h[1] = (_Float16)W[(2 * w + 1) * C_DIM + n];
        sWp[n * 32 + w] = pk.u;
    }
    __syncthreads();

    int wave   = t >> 5;
    int lane   = t & 31;
    int laneHi = lane >> 4;     // 0: lanes 0-15, 1: lanes 16-31
    int lane16 = lane & 15;
    int mtile  = wave >> 2;     // 0..1
    int ntile  = wave & 3;      // 0..3
    int row0   = blockIdx.x * 32 + mtile * 16;
    int nc     = ntile * 16 + lane16;

    int m = row0 + lane16;
    if (m >= N) m = N - 1;                     // clamp: EXEC must stay all-ones
    const float4* xr4 = (const float4*)(X + (size_t)m * C_DIM);

    v8f acc = {};
#pragma unroll
    for (int kt = 0; kt < 2; ++kt) {
        // --- A fragment (16x32 f16): a[0..15] <-> k = kt*32+laneHi*16+0..15 ---
        float4 c0 = xr4[kt * 8 + laneHi * 4];
        float4 c1 = xr4[kt * 8 + laneHi * 4 + 1];
        float4 c2 = xr4[kt * 8 + laneHi * 4 + 2];
        float4 c3 = xr4[kt * 8 + laneHi * 4 + 3];
        v16h a;
        a[0]  = (_Float16)c0.x; a[1]  = (_Float16)c0.y;
        a[2]  = (_Float16)c0.z; a[3]  = (_Float16)c0.w;
        a[4]  = (_Float16)c1.x; a[5]  = (_Float16)c1.y;
        a[6]  = (_Float16)c1.z; a[7]  = (_Float16)c1.w;
        a[8]  = (_Float16)c2.x; a[9]  = (_Float16)c2.y;
        a[10] = (_Float16)c2.z; a[11] = (_Float16)c2.w;
        a[12] = (_Float16)c3.x; a[13] = (_Float16)c3.y;
        a[14] = (_Float16)c3.z; a[15] = (_Float16)c3.w;
        // --- B fragment (32x16 f16): 8 contiguous k-pair dwords -> 2x ds_load_b128 ---
        const v16h b = *(const v16h*)&sWp[nc * 32 + kt * 16 + laneHi * 8];
        acc = __builtin_amdgcn_wmma_f32_16x16x32_f16(
            /*neg_a=*/false, a, /*neg_b=*/false, b,
            /*c_mod=*/(short)0, acc, /*reuse_a=*/false, /*reuse_b=*/false);
    }

    // --- D store: VGPR v -> row (v + laneHi*8), col lane16 ---
    if (row0 + 16 <= N) {                       // full tile: straight-line stores
        float* yo = Y + (size_t)(row0 + laneHi * 8) * C_DIM + nc;
#pragma unroll
        for (int v = 0; v < 8; ++v)
            yo[(size_t)v * C_DIM] = acc[v];
    } else {
#pragma unroll
        for (int v = 0; v < 8; ++v) {
            int mr = row0 + v + laneHi * 8;
            if (mr < N)
                Y[(size_t)mr * C_DIM + nc] = acc[v];
        }
    }
}

// ---------------- self-loop: agg = xw * dinv^2 (initializes agg), float4 ----------------
__global__ void stgcn_selfloop(const float* __restrict__ xw, const float* __restrict__ dinv,
                               float* __restrict__ agg, int total4) {
    int i = blockIdx.x * blockDim.x + threadIdx.x;
    if (i < total4) {
        float d = dinv[i >> 4];
        float c = d * d;
        float4 v = ((const float4*)xw)[i];
        v.x *= c; v.y *= c; v.z *= c; v.w *= c;
        ((float4*)agg)[i] = v;
    }
}

// ---------------- edge scatter: agg[dst] += xw[src] * dinv[src]*dinv[dst] ----------------
// 16 threads per edge, 4 features each: float4 gather + 4 float atomics.
__global__ void stgcn_scatter(const int* __restrict__ src, const int* __restrict__ dst,
                              const float* __restrict__ dinv, const float* __restrict__ xw,
                              float* __restrict__ agg, int total16) {
    int i = blockIdx.x * blockDim.x + threadIdx.x;
    if (i >= total16) return;
    int e = i >> 4;
    int f = (i & 15) << 2;
    int s = src[e], d = dst[e];
    float c = dinv[s] * dinv[d];
    float4 xv = *(const float4*)(xw + (size_t)s * C_DIM + f);
    float* ap = agg + (size_t)d * C_DIM + f;
    atomicAdd(ap + 0, xv.x * c);
    atomicAdd(ap + 1, xv.y * c);
    atomicAdd(ap + 2, xv.z * c);
    atomicAdd(ap + 3, xv.w * c);
}

// ---------------- bias + relu, in place, float4 ----------------
__global__ void stgcn_bias_relu(float* __restrict__ h, const float* __restrict__ b, int total4) {
    int i = blockIdx.x * blockDim.x + threadIdx.x;
    if (i < total4) {
        int f = (i & 15) << 2;
        float4 v = ((float4*)h)[i];
        v.x = fmaxf(v.x + b[f + 0], 0.0f);
        v.y = fmaxf(v.y + b[f + 1], 0.0f);
        v.z = fmaxf(v.z + b[f + 2], 0.0f);
        v.w = fmaxf(v.w + b[f + 3], 0.0f);
        ((float4*)h)[i] = v;
    }
}

// ---------------- per-edge prediction ----------------
__global__ void stgcn_pred(const int* __restrict__ src, const float* __restrict__ h,
                           const float* __restrict__ ea, const float* __restrict__ Wfc,
                           const float* __restrict__ bfc, float* __restrict__ pred, int E) {
    int e = blockIdx.x * blockDim.x + threadIdx.x;
    if (e >= E) return;
    const float4* hr = (const float4*)(h + (size_t)src[e] * C_DIM);
    float acc = 0.0f;
#pragma unroll
    for (int q = 0; q < 16; ++q) {
        float4 hv = hr[q];
        acc += hv.x * Wfc[4 * q] + hv.y * Wfc[4 * q + 1] +
               hv.z * Wfc[4 * q + 2] + hv.w * Wfc[4 * q + 3];
    }
    acc += ea[2 * e] * Wfc[64] + ea[2 * e + 1] * Wfc[65] + bfc[0];
    pred[e] = acc;
}

// ---------------- route segment sum: one block per route ----------------
__global__ __launch_bounds__(128) void stgcn_route_sum(const float* __restrict__ pred,
                                                       const int* __restrict__ ri,
                                                       float* __restrict__ out, int R) {
    int r = blockIdx.x;
    if (r >= R) return;
    int s = ri[2 * r], en = ri[2 * r + 1];
    float acc = 0.0f;
    for (int e = s + (int)threadIdx.x; e < en; e += 128)
        acc += pred[e];
#pragma unroll
    for (int off = 16; off > 0; off >>= 1)
        acc += __shfl_down(acc, off, 32);
    __shared__ float red[4];
    if ((threadIdx.x & 31) == 0) red[threadIdx.x >> 5] = acc;
    __syncthreads();
    if (threadIdx.x == 0)
        out[r] = red[0] + red[1] + red[2] + red[3];
}

extern "C" void kernel_launch(void* const* d_in, const int* in_sizes, int n_in,
                              void* d_out, int out_size, void* d_ws, size_t ws_size,
                              hipStream_t stream) {
    const float* x          = (const float*)d_in[0];
    const float* edge_attr  = (const float*)d_in[1];
    const float* W1         = (const float*)d_in[2];
    const float* b1         = (const float*)d_in[3];
    const float* W2         = (const float*)d_in[4];
    const float* b2         = (const float*)d_in[5];
    const float* Wfc        = (const float*)d_in[6];
    const float* bfc        = (const float*)d_in[7];
    const int*   edge_index = (const int*)d_in[8];
    const int*   route_idx  = (const int*)d_in[9];

    const int N = in_sizes[0] / C_DIM;
    const int E = in_sizes[8] / 2;
    const int R = in_sizes[9] / 2;
    const int* src = edge_index;
    const int* dst = edge_index + E;

    // workspace carve-out
    char* ws = (char*)d_ws;
    size_t off = 0;
    auto carve = [&](size_t bytes) -> void* {
        void* p = ws + off;
        off += (bytes + 255) & ~(size_t)255;
        return p;
    };
    float* dinv = (float*)carve((size_t)N * 4);
    float* bufA = (float*)carve((size_t)N * C_DIM * 4);   // xw
    float* bufB = (float*)carve((size_t)N * C_DIM * 4);   // agg / h
    float* pred = (float*)carve((size_t)E * 4);

    const int NT4  = N * (C_DIM / 4);   // float4 elements per node buffer
    const int ET16 = E * 16;            // scatter threads (4 features each)
    const int TB   = 256;

    // degree -> dinv
    stgcn_init_deg <<<(N + TB - 1) / TB, TB, 0, stream>>>(dinv, N);
    stgcn_count_deg<<<(E + TB - 1) / TB, TB, 0, stream>>>(dst, dinv, E);
    stgcn_dinv     <<<(N + TB - 1) / TB, TB, 0, stream>>>(dinv, N);

    // layer 1
    stgcn_gemm64_wmma<<<(N + 31) / 32, 256, 0, stream>>>(x, W1, bufA, N);
    stgcn_selfloop   <<<(NT4 + TB - 1) / TB, TB, 0, stream>>>(bufA, dinv, bufB, NT4);
    stgcn_scatter    <<<(ET16 + TB - 1) / TB, TB, 0, stream>>>(src, dst, dinv, bufA, bufB, ET16);
    stgcn_bias_relu  <<<(NT4 + TB - 1) / TB, TB, 0, stream>>>(bufB, b1, NT4);

    // layer 2 (h1 in bufB consumed by GEMM before bufB is overwritten)
    stgcn_gemm64_wmma<<<(N + 31) / 32, 256, 0, stream>>>(bufB, W2, bufA, N);
    stgcn_selfloop   <<<(NT4 + TB - 1) / TB, TB, 0, stream>>>(bufA, dinv, bufB, NT4);
    stgcn_scatter    <<<(ET16 + TB - 1) / TB, TB, 0, stream>>>(src, dst, dinv, bufA, bufB, ET16);
    stgcn_bias_relu  <<<(NT4 + TB - 1) / TB, TB, 0, stream>>>(bufB, b2, NT4);

    // edge scores + route reduction
    stgcn_pred     <<<(E + TB - 1) / TB, TB, 0, stream>>>(src, bufB, edge_attr, Wfc, bfc, pred, E);
    stgcn_route_sum<<<R, 128, 0, stream>>>(pred, route_idx, (float*)d_out, R);
}
